// BridgeModule_87084756894358
// MI455X (gfx1250) — matrix operand order repaired and checked by reading, not compiled, not run on hardware
//
#include <hip/hip_runtime.h>
#include <math.h>

// ---------------------------------------------------------------------------
// Problem constants (from the reference)
// ---------------------------------------------------------------------------
static constexpr int kB   = 8;
static constexpr int kSV  = 257;
static constexpr int kSQ  = 2048;
static constexpr int kDV  = 1024;
static constexpr int kDM  = 2304;
static constexpr int kH   = 8;
static constexpr int kDK  = 288;    // DM / H
static constexpr int kDF  = 9216;   // 4 * DM
static constexpr int kSVP = 288;    // SV padded to multiple of 32 for K-dim of attn@v

typedef __attribute__((ext_vector_type(16))) __bf16 v16bf;
typedef __attribute__((ext_vector_type(8)))  float  v8f;

__device__ __forceinline__ unsigned short f2bf(float f) {
  unsigned int u = __float_as_uint(f);
  u += 0x7fffu + ((u >> 16) & 1u);          // round-to-nearest-even
  return (unsigned short)(u >> 16);
}

union FragU { v16bf v; uint4 q[2]; };

// 32-bit DS address of a shared-memory location (flat shared pointers keep the
// LDS byte address in their low 32 bits on amdgcn).
__device__ __forceinline__ unsigned lds_addr32(const void* p) {
  return (unsigned)(unsigned long long)p;
}

// gfx1250 async DMA: global -> LDS, 16 bytes per lane, tracked by ASYNCcnt.
__device__ __forceinline__ void async_ld128(unsigned lds, const unsigned short* g) {
  asm volatile("global_load_async_to_lds_b128 %0, %1, off"
               :: "v"(lds), "v"(g) : "memory");
}
__device__ __forceinline__ void wait_async_le4() {
  asm volatile("s_wait_asynccnt 0x4" ::: "memory");
}
__device__ __forceinline__ void wait_async_0() {
  asm volatile("s_wait_asynccnt 0x0" ::: "memory");
}

// ---------------------------------------------------------------------------
// Generic bf16 WMMA GEMM:  C[m][n] = alpha * sum_k A[m][k] * Bt[n][k] (+ bias,
// gelu, residual).  A: [M x K] bf16 row-major (lda).  Bt: [N x K] bf16
// row-major (ldb) i.e. the weight matrix pre-transposed so K is contiguous.
// Batched over blockIdx.z via element strides sAz/sBz/sCz.
// Block = 256 threads = 8 wave32; block tile 128x128; wave tile 32x64
// (2x4 fragments of 16x16), K step 32.  LDS double-buffered, filled with
// GLOBAL_LOAD_ASYNC_TO_LDS_B128 so tile t+1 streams in under tile t's WMMAs.
// Out-of-range A/B rows are CLAMPED (not zeroed): they only produce C rows /
// cols that the store guard discards, and clamping keeps the staging loop
// divergence-free.
// ---------------------------------------------------------------------------
struct GemmP {
  const unsigned short* A;
  const unsigned short* Bm;
  long long sAz, sBz, sCz;      // per-z element strides (0 if unbatched)
  int M, N, K, lda, ldb;
  const float* bias;            // indexed by col, nullable
  float alpha;
  int gelu;                     // exact GELU on result
  const float* resid; int ldr;  // residual added (plain row*ldr+col), nullable
  float*          outF; int ldcF;              // f32 output (plain), nullable
  unsigned short* outH; int modeH; int ldcH;   // bf16 output, nullable
  int Sdim;                     // rows-per-batch for HEADS modes
};

__global__ __launch_bounds__(256) void gemm_wmma_bf16(GemmP p) {
  __shared__ uint4 sA[2][128 * 4];   // 2 x (128 rows x 32 bf16)
  __shared__ uint4 sB[2][128 * 4];

  const int tid  = threadIdx.x;
  const int lane = tid & 31;
  const int wid  = tid >> 5;
  const int z    = blockIdx.z;

  const unsigned short* Ag = p.A  + (long long)z * p.sAz;
  const unsigned short* Bg = p.Bm + (long long)z * p.sBz;

  const int tileM = blockIdx.y * 128;
  const int tileN = blockIdx.x * 128;
  const int wM = (wid >> 1) * 32;       // wave row offset in block tile
  const int wN = (wid & 1) * 64;        // wave col offset in block tile
  const int hi = lane >> 4;             // lane half (0/1)
  const int fr = lane & 15;             // row/col within fragment

  // Staging: 128x32 bf16 tile = 512 x 16B chunks; 2 chunks of A and B per
  // thread (rows r0 and r0+64).
  const int r0  = tid >> 2;
  const int r1  = r0 + 64;
  const int kc  = tid & 3;
  const int gA0 = (tileM + r0 < p.M) ? tileM + r0 : p.M - 1;
  const int gA1 = (tileM + r1 < p.M) ? tileM + r1 : p.M - 1;
  const int gB0 = (tileN + r0 < p.N) ? tileN + r0 : p.N - 1;
  const int gB1 = (tileN + r1 < p.N) ? tileN + r1 : p.N - 1;
  const unsigned short* a0 = Ag + (size_t)gA0 * p.lda + kc * 8;
  const unsigned short* a1 = Ag + (size_t)gA1 * p.lda + kc * 8;
  const unsigned short* b0 = Bg + (size_t)gB0 * p.ldb + kc * 8;
  const unsigned short* b1 = Bg + (size_t)gB1 * p.ldb + kc * 8;
  unsigned lA0[2], lA1[2], lB0[2], lB1[2];
#pragma unroll
  for (int b = 0; b < 2; ++b) {
    lA0[b] = lds_addr32(&sA[b][r0 * 4 + kc]);
    lA1[b] = lds_addr32(&sA[b][r1 * 4 + kc]);
    lB0[b] = lds_addr32(&sB[b][r0 * 4 + kc]);
    lB1[b] = lds_addr32(&sB[b][r1 * 4 + kc]);
  }

  const v8f vzero = {0.f, 0.f, 0.f, 0.f, 0.f, 0.f, 0.f, 0.f};
  v8f acc[2][4];
#pragma unroll
  for (int i = 0; i < 2; ++i)
#pragma unroll
    for (int j = 0; j < 4; ++j) acc[i][j] = vzero;

  const int nT = p.K >> 5;              // K is always a multiple of 32
  auto issue = [&](int t) {
    const int b  = t & 1;
    const int kk = t << 5;              // element offset along K
    async_ld128(lA0[b], a0 + kk);
    async_ld128(lA1[b], a1 + kk);
    async_ld128(lB0[b], b0 + kk);
    async_ld128(lB1[b], b1 + kk);
  };

  issue(0);
  for (int t = 0; t < nT; ++t) {
    __syncthreads();                    // everyone done reading buf[(t+1)&1]
    if (t + 1 < nT) {
      issue(t + 1);                     // prefetch next tile (other buffer)
      wait_async_le4();                 // tile t complete; t+1 still in flight
    } else {
      wait_async_0();
    }
    __syncthreads();                    // tile t visible to all waves

    const uint4* cA = sA[t & 1];
    const uint4* cB = sB[t & 1];

    // A fragment (16x32, ISA 7.12.2): lanes 0-15 hold K 0-7 & 16-23,
    // lanes 16-31 hold K 8-15 & 24-31 for row M = lane&15.
    v16bf af[2];
#pragma unroll
    for (int mi = 0; mi < 2; ++mi) {
      const uint4* rp = &cA[(wM + mi * 16 + fr) * 4];
      FragU f; f.q[0] = rp[hi]; f.q[1] = rp[2 + hi];
      af[mi] = f.v;
    }
    // B fragment (32x16): lanes 0-15 hold K 0-7 & 16-23, lanes 16-31 hold
    // K 8-15 & 24-31 for col N = lane&15.
    v16bf bfr[4];
#pragma unroll
    for (int ni = 0; ni < 4; ++ni) {
      const uint4* rp = &cB[(wN + ni * 16 + fr) * 4];
      FragU f; f.q[0] = rp[hi]; f.q[1] = rp[2 + hi];
      bfr[ni] = f.v;
    }
#pragma unroll
    for (int mi = 0; mi < 2; ++mi)
#pragma unroll
      for (int ni = 0; ni < 4; ++ni)
        acc[mi][ni] = __builtin_amdgcn_wmma_f32_16x16x32_bf16(
            false, af[mi], false, bfr[ni], (short)0, acc[mi][ni], false, false);
  }

  // Epilogue.  C layout: VGPR r, lanes 0-15 -> M=r, lanes 16-31 -> M=r+8,
  // N = lane&15.
#pragma unroll
  for (int mi = 0; mi < 2; ++mi) {
#pragma unroll
    for (int ni = 0; ni < 4; ++ni) {
      int col = tileN + wN + ni * 16 + fr;
      if (col >= p.N) continue;
      float bv = p.bias ? p.bias[col] : 0.f;
#pragma unroll
      for (int r = 0; r < 8; ++r) {
        int row = tileM + wM + mi * 16 + r + (hi << 3);
        if (row >= p.M) continue;
        float v = acc[mi][ni][r] * p.alpha + bv;
        if (p.gelu) v = 0.5f * v * (1.0f + erff(v * 0.70710678118654752f));
        if (p.resid) v += p.resid[(size_t)row * p.ldr + col];
        if (p.outF)
          p.outF[(long long)z * p.sCz + (size_t)row * p.ldcF + col] = v;
        if (p.outH) {
          size_t off;
          if (p.modeH == 0) {                 // plain [M x ldcH]
            off = (size_t)row * p.ldcH + col;
          } else if (p.modeH == 1) {          // [B,H,S,DK] (q / k)
            int b = row / p.Sdim, s = row - b * p.Sdim;
            int h = col / kDK,    d = col - h * kDK;
            off = (((size_t)(b * kH + h)) * p.Sdim + s) * kDK + d;
          } else if (p.modeH == 2) {          // [B,H,DK,SVP] (v transposed)
            int b = row / p.Sdim, s = row - b * p.Sdim;
            int h = col / kDK,    d = col - h * kDK;
            off = (((size_t)(b * kH + h)) * kDK + d) * kSVP + s;
          } else {                            // ctx scatter: z=(b,h), row=s, col=d
            int b = z >> 3, h = z & 7;
            off = ((size_t)b * kSQ + row) * kDM + (size_t)h * kDK + col;
          }
          p.outH[off] = f2bf(v);
        }
      }
    }
  }
}

// ---------------------------------------------------------------------------
// LayerNorm over last dim D (f32 in -> bf16 out), one block per row.
// ---------------------------------------------------------------------------
__global__ __launch_bounds__(256) void layernorm_bf16(
    const float* __restrict__ x, const float* __restrict__ w,
    const float* __restrict__ b, unsigned short* __restrict__ out, int D) {
  __shared__ float s1[256], s2[256];
  const int tid = threadIdx.x;
  const size_t base = (size_t)blockIdx.x * D;
  float a1 = 0.f, a2 = 0.f;
  for (int c = tid; c < D; c += 256) {
    float v = x[base + c];
    a1 += v; a2 += v * v;
  }
  s1[tid] = a1; s2[tid] = a2;
  __syncthreads();
  for (int st = 128; st > 0; st >>= 1) {
    if (tid < st) { s1[tid] += s1[tid + st]; s2[tid] += s2[tid + st]; }
    __syncthreads();
  }
  float mean = s1[0] / D;
  float var  = s2[0] / D - mean * mean;
  float inv  = rsqrtf(var + 1e-5f);
  for (int c = tid; c < D; c += 256)
    out[base + c] = f2bf((x[base + c] - mean) * inv * w[c] + b[c]);
}

// ---------------------------------------------------------------------------
// Softmax over SV=257 scores (f32) -> bf16 probs, pad cols [257,288) = 0.
// One wave32 per row.
// ---------------------------------------------------------------------------
__global__ __launch_bounds__(256) void softmax_bf16(
    const float* __restrict__ sc, unsigned short* __restrict__ attn,
    long long rows) {
  const int lane = threadIdx.x & 31;
  long long row = (long long)blockIdx.x * 8 + (threadIdx.x >> 5);
  if (row >= rows) return;
  const float* p = sc + row * kSVP;
  float m = -3.0e38f;
  for (int c = lane; c < kSV; c += 32) m = fmaxf(m, p[c]);
  for (int o = 16; o > 0; o >>= 1) m = fmaxf(m, __shfl_xor(m, o, 32));
  float s = 0.f;
  for (int c = lane; c < kSV; c += 32) s += __expf(p[c] - m);
  for (int o = 16; o > 0; o >>= 1) s += __shfl_xor(s, o, 32);
  float inv = 1.0f / s;
  unsigned short* q = attn + row * kSVP;
  for (int c = lane; c < kSVP; c += 32)
    q[c] = (c < kSV) ? f2bf(__expf(p[c] - m) * inv) : (unsigned short)0;
}

// ---------------------------------------------------------------------------
// f32 -> bf16 converts (plain and transposed-for-weights)
// ---------------------------------------------------------------------------
__global__ void cvt_bf16(const float* __restrict__ in,
                         unsigned short* __restrict__ out, long long n) {
  long long i = (long long)blockIdx.x * 256 + threadIdx.x;
  if (i < n) out[i] = f2bf(in[i]);
}

__global__ void cvt_bf16_T(const float* __restrict__ in,
                           unsigned short* __restrict__ out, int K, int N) {
  long long i = (long long)blockIdx.x * 256 + threadIdx.x;
  if (i < (long long)K * N) {
    int k = (int)(i / N);
    int n = (int)(i - (long long)k * N);
    out[(size_t)n * K + k] = f2bf(in[i]);   // [K x N] -> [N x K]
  }
}

// ---------------------------------------------------------------------------
// Host launcher
// ---------------------------------------------------------------------------
extern "C" void kernel_launch(void* const* d_in, const int* in_sizes, int n_in,
                              void* d_out, int out_size, void* d_ws, size_t ws_size,
                              hipStream_t stream) {
  (void)in_sizes; (void)n_in; (void)out_size; (void)ws_size;

  const float* vision = (const float*)d_in[0];
  const float* text   = (const float*)d_in[1];
  const float* vp_w   = (const float*)d_in[2];
  const float* vp_b   = (const float*)d_in[3];
  const float* ln1w   = (const float*)d_in[4];
  const float* ln1b   = (const float*)d_in[5];
  const float* ln2w   = (const float*)d_in[6];
  const float* ln2b   = (const float*)d_in[7];
  const float* wq_w   = (const float*)d_in[8];
  const float* wq_b   = (const float*)d_in[9];
  const float* wk_w   = (const float*)d_in[10];
  const float* wk_b   = (const float*)d_in[11];
  const float* wv_w   = (const float*)d_in[12];
  const float* wv_b   = (const float*)d_in[13];
  const float* wo_w   = (const float*)d_in[14];
  const float* wo_b   = (const float*)d_in[15];
  const float* f1_w   = (const float*)d_in[16];
  const float* f1_b   = (const float*)d_in[17];
  const float* f2_w   = (const float*)d_in[18];
  const float* f2_b   = (const float*)d_in[19];

  char* ws = (char*)d_ws;
  size_t off = 0;
  auto alloc = [&](size_t bytes) -> void* {
    void* p = ws + off;
    off = (off + bytes + 255) & ~(size_t)255;
    return p;
  };

  const long long MV = (long long)kB * kSV;   // 2056 vision rows
  const long long MT = (long long)kB * kSQ;   // 16384 text rows

  unsigned short* vis_h = (unsigned short*)alloc((size_t)MV * kDV * 2);
  unsigned short* vp_t  = (unsigned short*)alloc((size_t)kDM * kDV * 2);
  unsigned short* wq_t  = (unsigned short*)alloc((size_t)kDM * kDM * 2);
  unsigned short* wk_t  = (unsigned short*)alloc((size_t)kDM * kDM * 2);
  unsigned short* wv_t  = (unsigned short*)alloc((size_t)kDM * kDM * 2);
  unsigned short* wo_t  = (unsigned short*)alloc((size_t)kDM * kDM * 2);
  unsigned short* f1_t  = (unsigned short*)alloc((size_t)kDF * kDM * 2);
  unsigned short* f2_t  = (unsigned short*)alloc((size_t)kDM * kDF * 2);
  unsigned short* nt_h  = (unsigned short*)alloc((size_t)MT * kDM * 2);              // LN1 out; reused as ctx
  unsigned short* pv_h  = (unsigned short*)alloc((size_t)MV * kDM * 2);
  unsigned short* q_h   = (unsigned short*)alloc((size_t)kB * kH * kSQ * kDK * 2);
  unsigned short* k_h   = (unsigned short*)alloc((size_t)kB * kH * kSV * kDK * 2);
  unsigned short* vt_h  = (unsigned short*)alloc((size_t)kB * kH * kDK * kSVP * 2);
  float*          sc_f  = (float*)alloc((size_t)kB * kH * kSQ * kSVP * 4);           // scores; reused as x
  unsigned short* at_h  = (unsigned short*)alloc((size_t)kB * kH * kSQ * kSVP * 2);  // attn; reused as nx
  unsigned short* h_h   = (unsigned short*)alloc((size_t)MT * kDF * 2);

  unsigned short* ctx_h = nt_h;   // nt dead after q GEMM
  float*          x_f   = sc_f;   // scores dead after softmax
  unsigned short* nx_h  = at_h;   // attn dead after ctx GEMM

  // ---- converts + LN1 ----
  {
    long long n = MV * kDV;
    cvt_bf16<<<dim3((unsigned)((n + 255) / 256)), dim3(256), 0, stream>>>(vision, vis_h, n);
  }
  auto cvtT = [&](const float* src, unsigned short* dst, int K, int N) {
    long long n = (long long)K * N;
    cvt_bf16_T<<<dim3((unsigned)((n + 255) / 256)), dim3(256), 0, stream>>>(src, dst, K, N);
  };
  cvtT(vp_w, vp_t, kDV, kDM);
  cvtT(wq_w, wq_t, kDM, kDM);
  cvtT(wk_w, wk_t, kDM, kDM);
  cvtT(wv_w, wv_t, kDM, kDM);
  cvtT(wo_w, wo_t, kDM, kDM);
  cvtT(f1_w, f1_t, kDM, kDF);
  cvtT(f2_w, f2_t, kDF, kDM);

  layernorm_bf16<<<dim3((unsigned)MT), dim3(256), 0, stream>>>(text, ln1w, ln1b, nt_h, kDM);
  hipMemsetAsync(vt_h, 0, (size_t)kB * kH * kDK * kSVP * 2, stream);  // zero K-pad rows

  auto launch = [&](const GemmP& p, int gx, int gy, int gz) {
    gemm_wmma_bf16<<<dim3(gx, gy, gz), dim3(256), 0, stream>>>(p);
  };

  { // pv = vision @ vp_w + vp_b  -> bf16 [MV x DM]
    GemmP p{}; p.A = vis_h; p.Bm = vp_t; p.M = (int)MV; p.N = kDM; p.K = kDV;
    p.lda = kDV; p.ldb = kDV; p.bias = vp_b; p.alpha = 1.f;
    p.outH = pv_h; p.modeH = 0; p.ldcH = kDM;
    launch(p, (kDM + 127) / 128, ((int)MV + 127) / 128, 1);
  }
  { // q = nt @ wq + b -> [B,H,SQ,DK]
    GemmP p{}; p.A = nt_h; p.Bm = wq_t; p.M = (int)MT; p.N = kDM; p.K = kDM;
    p.lda = kDM; p.ldb = kDM; p.bias = wq_b; p.alpha = 1.f;
    p.outH = q_h; p.modeH = 1; p.Sdim = kSQ;
    launch(p, kDM / 128, (int)MT / 128, 1);
  }
  { // k = pv @ wk + b -> [B,H,SV,DK]
    GemmP p{}; p.A = pv_h; p.Bm = wk_t; p.M = (int)MV; p.N = kDM; p.K = kDM;
    p.lda = kDM; p.ldb = kDM; p.bias = wk_b; p.alpha = 1.f;
    p.outH = k_h; p.modeH = 1; p.Sdim = kSV;
    launch(p, kDM / 128, ((int)MV + 127) / 128, 1);
  }
  { // v = pv @ wv + b -> transposed [B,H,DK,SVP]
    GemmP p{}; p.A = pv_h; p.Bm = wv_t; p.M = (int)MV; p.N = kDM; p.K = kDM;
    p.lda = kDM; p.ldb = kDM; p.bias = wv_b; p.alpha = 1.f;
    p.outH = vt_h; p.modeH = 2; p.Sdim = kSV;
    launch(p, kDM / 128, ((int)MV + 127) / 128, 1);
  }
  { // scores[z] = (q[z] @ k[z]^T) / sqrt(DK), z = b*H+h
    GemmP p{}; p.A = q_h; p.Bm = k_h;
    p.sAz = (long long)kSQ * kDK; p.sBz = (long long)kSV * kDK;
    p.sCz = (long long)kSQ * kSVP;
    p.M = kSQ; p.N = kSV; p.K = kDK; p.lda = kDK; p.ldb = kDK;
    p.alpha = 0.05892556509887896f;   // 1/sqrt(288)
    p.outF = sc_f; p.ldcF = kSVP;
    launch(p, (kSV + 127) / 128, kSQ / 128, kB * kH);
  }
  softmax_bf16<<<dim3((unsigned)((long long)kB * kH * kSQ / 8)), dim3(256), 0, stream>>>(
      sc_f, at_h, (long long)kB * kH * kSQ);
  { // ctx[z] = attn[z] @ v[z]  (K = SVP with zero pad), scatter to [B*SQ, DM]
    GemmP p{}; p.A = at_h; p.Bm = vt_h;
    p.sAz = (long long)kSQ * kSVP; p.sBz = (long long)kDK * kSVP; p.sCz = 0;
    p.M = kSQ; p.N = kDK; p.K = kSVP; p.lda = kSVP; p.ldb = kSVP; p.alpha = 1.f;
    p.outH = ctx_h; p.modeH = 3;
    launch(p, (kDK + 127) / 128, kSQ / 128, kB * kH);
  }
  { // x = text + ctx @ wo + b  -> f32
    GemmP p{}; p.A = ctx_h; p.Bm = wo_t; p.M = (int)MT; p.N = kDM; p.K = kDM;
    p.lda = kDM; p.ldb = kDM; p.bias = wo_b; p.alpha = 1.f;
    p.resid = text; p.ldr = kDM;
    p.outF = x_f; p.ldcF = kDM;
    launch(p, kDM / 128, (int)MT / 128, 1);
  }
  layernorm_bf16<<<dim3((unsigned)MT), dim3(256), 0, stream>>>(x_f, ln2w, ln2b, nx_h, kDM);
  { // h = gelu(nx @ f1 + b) -> bf16 [MT x DF]
    GemmP p{}; p.A = nx_h; p.Bm = f1_t; p.M = (int)MT; p.N = kDF; p.K = kDM;
    p.lda = kDM; p.ldb = kDM; p.bias = f1_b; p.alpha = 1.f; p.gelu = 1;
    p.outH = h_h; p.modeH = 0; p.ldcH = kDF;
    launch(p, kDF / 128, (int)MT / 128, 1);
  }
  { // out = x + h @ f2 + b -> f32
    GemmP p{}; p.A = h_h; p.Bm = f2_t; p.M = (int)MT; p.N = kDM; p.K = kDF;
    p.lda = kDF; p.ldb = kDF; p.bias = f2_b; p.alpha = 1.f;
    p.resid = x_f; p.ldr = kDM;
    p.outF = (float*)d_out; p.ldcF = kDM;
    launch(p, kDM / 128, (int)MT / 128, 1);
  }
}
